// PathDiscovery_14164802142496
// MI455X (gfx1250) — matrix operand order repaired and checked by reading, not compiled, not run on hardware
//
#include <hip/hip_runtime.h>
#include <stdint.h>

// Problem constants (match reference)
#define BB 8
#define TT 12
#define NN 4096
#define FF 64
#define HH 64
#define KK 8
#define LL 5
#define ALPHA_C 0.15f

typedef __attribute__((ext_vector_type(16))) __bf16 v16bf;
typedef __attribute__((ext_vector_type(8)))  float  v8f;

// ---------------------------------------------------------------- helpers ---
// Native f32 -> bf16 conversion (RNE); lowers to v_cvt_*bf16_f32 on gfx1250.
__device__ __forceinline__ __bf16 f2bf(float f) { return (__bf16)f; }

// A fragment, 16x32 bf16 (row-major fp32 source).
// lane<16: M=lane, K = k0+{0..7} in v0..3(lo/hi pairs), k0+{16..23} in v4..7
// lane>=16: M=lane-16, K = k0+{8..15}, k0+{24..31}
__device__ __forceinline__ v16bf load_a_frag(const float* __restrict__ p, int ld,
                                             int lane, int k0) {
  const float* rp = p + (size_t)(lane & 15) * ld + k0 + ((lane >= 16) ? 8 : 0);
  v16bf v;
#pragma unroll
  for (int i = 0; i < 8; i++) v[i] = f2bf(rp[i]);
#pragma unroll
  for (int i = 0; i < 8; i++) v[8 + i] = f2bf(rp[16 + i]);
  return v;
}

// B fragment for D = A x B with B = W^T, W row-major (H x F): column n of B is
// row n of W.  lane<16: col=lane, K=k0+{0..7},{16..23}; lane>=16: col=lane-16, K=k0+{8..15},{24..31}
__device__ __forceinline__ v16bf load_b_frag(const float* __restrict__ w, int ld,
                                             int lane, int n0, int k0) {
  const float* rp = w + (size_t)(n0 + (lane & 15)) * ld + k0 + ((lane >= 16) ? 8 : 0);
  v16bf v;
#pragma unroll
  for (int i = 0; i < 8; i++) v[i] = f2bf(rp[i]);
#pragma unroll
  for (int i = 0; i < 8; i++) v[8 + i] = f2bf(rp[16 + i]);
  return v;
}

// ------------------------------------------------------ JAX Threefry2x32 ----
__device__ __forceinline__ unsigned rotl32(unsigned x, int r) {
  return (x << r) | (x >> (32 - r));
}
__device__ __forceinline__ void tf2x32(unsigned k0, unsigned k1,
                                       unsigned c0, unsigned c1,
                                       unsigned& o0, unsigned& o1) {
  const unsigned ks2 = k0 ^ k1 ^ 0x1BD11BDAu;
  unsigned x0 = c0 + k0, x1 = c1 + k1;
  const int ra[4] = {13, 15, 26, 6};
  const int rb[4] = {17, 29, 16, 24};
#pragma unroll
  for (int i = 0; i < 4; i++) { x0 += x1; x1 = rotl32(x1, ra[i]); x1 ^= x0; }
  x0 += k1; x1 += ks2 + 1u;
#pragma unroll
  for (int i = 0; i < 4; i++) { x0 += x1; x1 = rotl32(x1, rb[i]); x1 ^= x0; }
  x0 += ks2; x1 += k0 + 2u;
#pragma unroll
  for (int i = 0; i < 4; i++) { x0 += x1; x1 = rotl32(x1, ra[i]); x1 ^= x0; }
  x0 += k0; x1 += k1 + 3u;
#pragma unroll
  for (int i = 0; i < 4; i++) { x0 += x1; x1 = rotl32(x1, rb[i]); x1 ^= x0; }
  x0 += k1; x1 += ks2 + 4u;
#pragma unroll
  for (int i = 0; i < 4; i++) { x0 += x1; x1 = rotl32(x1, ra[i]); x1 ^= x0; }
  x0 += ks2; x1 += k0 + 5u;
  o0 = x0; o1 = x1;
}
// j-th subkey of jax.random.split(key, n)
__device__ __forceinline__ void split_key(unsigned k0, unsigned k1, int n, int j,
                                          unsigned& r0, unsigned& r1) {
  unsigned f[2];
#pragma unroll
  for (int s = 0; s < 2; s++) {
    int i = 2 * j + s;
    int p = (i < n) ? i : i - n;
    unsigned a, b;
    tf2x32(k0, k1, (unsigned)p, (unsigned)(n + p), a, b);
    f[s] = (i < n) ? a : b;
  }
  r0 = f[0]; r1 = f[1];
}
__device__ __forceinline__ float bits_to_unit(unsigned bits) {
  union { unsigned u; float f; } c; c.u = (bits >> 9) | 0x3f800000u;
  return c.f - 1.0f;
}
__device__ __forceinline__ float uniform_scalar(unsigned k0, unsigned k1) {
  unsigned o0, o1; tf2x32(k0, k1, 0u, 0u, o0, o1);
  return bits_to_unit(o0);
}
// Gumbel sample j of a length-NN draw under key (k0,k1)
__device__ __forceinline__ float gumbel_at(unsigned k0, unsigned k1, int j) {
  const int half = NN / 2;
  int p = (j < half) ? j : j - half;
  unsigned o0, o1; tf2x32(k0, k1, (unsigned)p, (unsigned)(half + p), o0, o1);
  unsigned bits = (j < half) ? o0 : o1;
  const float tiny = 1.17549435e-38f;
  float u = bits_to_unit(bits) * (1.0f - tiny) + tiny;
  u = fmaxf(u, tiny);
  return -logf(-logf(u));
}

// =================================================================== K1 =====
// fchange[b][n] = mean_{t,f} |nf[b,t,n,f] - nf[b,t-1,n,f]|   (one wave/(b,n))
__global__ void __launch_bounds__(256)
k_fchange(const float* __restrict__ nf, float* __restrict__ fch) {
  const int wave = (int)((blockIdx.x * blockDim.x + threadIdx.x) >> 5);
  const int lane = threadIdx.x & 31;
  if (wave >= BB * NN) return;
  const int b = wave >> 12, n = wave & (NN - 1);
  const float* base = nf + (((size_t)b * TT) * NN + n) * FF + lane * 2;
  const size_t ts = (size_t)NN * FF;
  float2 prev = *(const float2*)(base);
  float acc = 0.f;
#pragma unroll
  for (int t = 1; t < TT; t++) {
    if (t + 1 < TT) __builtin_prefetch(base + (size_t)(t + 1) * ts, 0, 0);
    float2 c = *(const float2*)(base + (size_t)t * ts);
    acc += fabsf(c.x - prev.x) + fabsf(c.y - prev.y);
    prev = c;
  }
#pragma unroll
  for (int o = 16; o >= 1; o >>= 1) acc += __shfl_xor(acc, o, 32);
  if (lane == 0) fch[wave] = acc * (1.0f / ((TT - 1) * FF));
}

// =================================================================== K2 =====
// scaled[b,n] = (relu(nf_last @ W1^T + b1) @ W2^T + b2) * fchange  via WMMA bf16
__global__ void __launch_bounds__(128)
k_srcmlp(const float* __restrict__ nf, const float* __restrict__ W1,
         const float* __restrict__ b1, const float* __restrict__ W2,
         const float* __restrict__ b2, const float* __restrict__ fch,
         float* __restrict__ scaled) {
  const int wave = (int)((blockIdx.x * blockDim.x + threadIdx.x) >> 5);
  const int lane = threadIdx.x & 31;
  const int gRow = wave * 16;              // tile base row in [0, B*N)
  const int b = gRow >> 12;
  const int row = gRow & (NN - 1);
  const float* A = nf + ((size_t)(b * TT + (TT - 1)) * NN + row) * FF;

  v8f acc0 = {}, acc1 = {}, acc2 = {}, acc3 = {};
#pragma unroll
  for (int ks = 0; ks < 2; ks++) {
    const int k0 = ks * 32;
    v16bf a = load_a_frag(A, FF, lane, k0);
    v16bf w0 = load_b_frag(W1, FF, lane, 0, k0);
    acc0 = __builtin_amdgcn_wmma_f32_16x16x32_bf16(false, a, false, w0, (short)0, acc0, false, false);
    v16bf w1 = load_b_frag(W1, FF, lane, 16, k0);
    acc1 = __builtin_amdgcn_wmma_f32_16x16x32_bf16(false, a, false, w1, (short)0, acc1, false, false);
    v16bf w2 = load_b_frag(W1, FF, lane, 32, k0);
    acc2 = __builtin_amdgcn_wmma_f32_16x16x32_bf16(false, a, false, w2, (short)0, acc2, false, false);
    v16bf w3 = load_b_frag(W1, FF, lane, 48, k0);
    acc3 = __builtin_amdgcn_wmma_f32_16x16x32_bf16(false, a, false, w3, (short)0, acc3, false, false);
  }

  // fused epilogue: relu, dot with W_sp2, +b2, * fchange
  const int col = lane & 15;
  float part[8];
#pragma unroll
  for (int r = 0; r < 8; r++) {
    float s = 0.f, h;
    h = acc0[r] + b1[col];        s += fmaxf(h, 0.f) * W2[col];
    h = acc1[r] + b1[16 + col];   s += fmaxf(h, 0.f) * W2[16 + col];
    h = acc2[r] + b1[32 + col];   s += fmaxf(h, 0.f) * W2[32 + col];
    h = acc3[r] + b1[48 + col];   s += fmaxf(h, 0.f) * W2[48 + col];
    part[r] = s;
  }
#pragma unroll
  for (int r = 0; r < 8; r++) {
#pragma unroll
    for (int o = 8; o >= 1; o >>= 1) part[r] += __shfl_xor(part[r], o, 16);
  }
  if (col == 0) {
    const int mbase = gRow + ((lane >= 16) ? 8 : 0);
    const float bias = b2[0];
#pragma unroll
    for (int r = 0; r < 8; r++) {
      int m = mbase + r;
      scaled[m] = (part[r] + bias) * fch[m];
    }
  }
}

// =================================================================== K3 =====
// per-batch top-K indices of scaled logits (softmax is monotone)
__global__ void __launch_bounds__(256)
k_topk(const float* __restrict__ scaled, int* __restrict__ src) {
  const int b = blockIdx.x, tid = threadIdx.x;
  __shared__ float sv[256];
  __shared__ int si[256];
  __shared__ int sel[KK];
  const float* s = scaled + (size_t)b * NN;
  for (int k = 0; k < KK; k++) {
    float bv = -__builtin_inff(); int bi = 0x7fffffff;
    for (int j = tid; j < NN; j += 256) {
      bool taken = false;
      for (int q = 0; q < k; q++) taken |= (sel[q] == j);
      if (taken) continue;
      float v = s[j];
      if (v > bv) { bv = v; bi = j; }
    }
    sv[tid] = bv; si[tid] = bi;
    __syncthreads();
    for (int o = 128; o >= 1; o >>= 1) {
      if (tid < o) {
        float v2 = sv[tid + o]; int i2 = si[tid + o];
        if (v2 > sv[tid] || (v2 == sv[tid] && i2 < si[tid])) { sv[tid] = v2; si[tid] = i2; }
      }
      __syncthreads();
    }
    if (tid == 0) { sel[k] = si[0]; src[b * KK + k] = si[0]; }
    __syncthreads();
  }
}

// =================================================================== K4 =====
// biased random walks; Threefry RNG stream mirrors jax.random
__global__ void __launch_bounds__(256)
k_walk(const float* __restrict__ adj, const float* __restrict__ nf,
       const float* __restrict__ fch, const int* __restrict__ src,
       int* __restrict__ paths) {
  const int w = blockIdx.x;            // walk id = b*K + k
  const int b = w >> 3;
  const int tid = threadIdx.x;
  __shared__ float diff[NN];
  __shared__ unsigned char visited[NN];
  __shared__ float curf[FF];
  __shared__ float red[256];
  __shared__ int redi[256];
  __shared__ int s_cur, s_done;
  __shared__ float s_sumM, s_sumP;

  const float* feats = nf + (size_t)(b * TT + (TT - 1)) * NN * FF;
  const float* adjb  = adj + (size_t)b * NN * NN;
  const float* fc    = fch + (size_t)b * NN;

  // acc_bias denominator
  float ls = 0.f;
  for (int j = tid; j < NN; j += 256) ls += fc[j];
  red[tid] = ls; __syncthreads();
  for (int o = 128; o >= 1; o >>= 1) { if (tid < o) red[tid] += red[tid + o]; __syncthreads(); }
  const float accsum = red[0] + 1e-8f;
  __syncthreads();

  const int source = src[w];
  for (int j = tid; j < NN; j += 256) visited[j] = 0;
  if (tid < LL) paths[w * LL + tid] = source;
  if (tid == 0) { s_cur = source; s_done = 0; visited[source] = 1; }
  __syncthreads();

  // walk key: split(key(42), 64)[w]; step keys: split(wk, 4)
  unsigned wk0, wk1;
  split_key(0u, 42u, BB * KK, w, wk0, wk1);

  for (int step = 0; step < LL - 1; step++) {
    unsigned sk0, sk1, ku0, ku1, kc0, kc1;
    split_key(wk0, wk1, LL - 1, step, sk0, sk1);
    split_key(sk0, sk1, 2, 0, ku0, ku1);
    split_key(sk0, sk1, 2, 1, kc0, kc1);

    const int cur = s_cur;
    if (tid < FF) curf[tid] = feats[(size_t)cur * FF + tid];
    __syncthreads();

    // pass 1: L2 feature distances + max
    float lmax = 0.f;
    for (int j = tid; j < NN; j += 256) {
      const float* fj = feats + (size_t)j * FF;
      float ss = 0.f;
#pragma unroll 8
      for (int f = 0; f < FF; f++) { float d = fj[f] - curf[f]; ss += d * d; }
      float dv = sqrtf(ss);
      diff[j] = dv;
      lmax = fmaxf(lmax, dv);
    }
    red[tid] = lmax; __syncthreads();
    for (int o = 128; o >= 1; o >>= 1) { if (tid < o) red[tid] = fmaxf(red[tid], red[tid + o]); __syncthreads(); }
    const float m = red[0];
    __syncthreads();

    // pass 2: probs, sums, Gumbel-argmax (categorical)
    float lsm = 0.f, lsp = 0.f;
    float bv = -__builtin_inff(); int bi = 0x7fffffff;
    const float* arow = adjb + (size_t)cur * NN;
    for (int j = tid; j < NN; j += 256) {
      float g = (m > 0.f) ? diff[j] / fmaxf(m, 1e-38f) : 0.f;
      float mk = visited[j] ? 0.f : arow[j];
      float pr = mk * (1.f + fc[j] / accsum) * (1.f + g);
      lsm += mk; lsp += pr;
      float lg = (pr > 0.f) ? logf(fmaxf(pr, 1e-38f)) : -__builtin_inff();
      float t = lg + gumbel_at(kc0, kc1, j);
      if (t > bv || (t == bv && j < bi)) { bv = t; bi = j; }
    }
    red[tid] = lsm; __syncthreads();
    for (int o = 128; o >= 1; o >>= 1) { if (tid < o) red[tid] += red[tid + o]; __syncthreads(); }
    if (tid == 0) s_sumM = red[0];
    __syncthreads();
    red[tid] = lsp; __syncthreads();
    for (int o = 128; o >= 1; o >>= 1) { if (tid < o) red[tid] += red[tid + o]; __syncthreads(); }
    if (tid == 0) s_sumP = red[0];
    __syncthreads();
    red[tid] = bv; redi[tid] = bi; __syncthreads();
    for (int o = 128; o >= 1; o >>= 1) {
      if (tid < o) {
        float v2 = red[tid + o]; int i2 = redi[tid + o];
        if (v2 > red[tid] || (v2 == red[tid] && i2 < redi[tid])) { red[tid] = v2; redi[tid] = i2; }
      }
      __syncthreads();
    }
    if (tid == 0) {
      int sampled = (redi[0] == 0x7fffffff) ? 0 : redi[0];
      bool dead = (s_sumM <= 0.f) || (s_sumP <= 0.f);
      bool tele = uniform_scalar(ku0, ku1) < ALPHA_C;
      int nxt = tele ? source : sampled;
      bool write = (!s_done) && (tele || !dead);
      if (write) {
        paths[w * LL + step + 1] = nxt;
        s_cur = nxt;
        visited[nxt] = 1;
      }
      if (!tele && dead) s_done = 1;
    }
    __syncthreads();
  }
}

// =================================================================== K5 =====
// gather path features + fwd GRU (5 steps) + 1-step bwd GRU; emit pf (64,128)
__global__ void __launch_bounds__(64)
k_gru(const float* __restrict__ nf, const int* __restrict__ paths,
      const float* __restrict__ Wihf, const float* __restrict__ Whhf,
      const float* __restrict__ bihf, const float* __restrict__ bhhf,
      const float* __restrict__ Wihb, const float* __restrict__ Whhb,
      const float* __restrict__ bihb, const float* __restrict__ bhhb,
      float* __restrict__ pf, float* __restrict__ out_pf) {
  const int s = blockIdx.x;            // sequence id = b*K + k
  const int b = s >> 3;
  const int h = threadIdx.x;
  __shared__ float xt[FF];
  __shared__ float hp[HH];
  const float* feats = nf + (size_t)(b * TT + (TT - 1)) * NN * FF;

  hp[h] = 0.f; __syncthreads();
  float hreg = 0.f;
  for (int t = 0; t < LL; t++) {
    const int node = paths[s * LL + t];
    xt[h] = feats[(size_t)node * FF + h];
    __syncthreads();
    float gir = bihf[h], giz = bihf[HH + h], gin = bihf[2 * HH + h];
    for (int f = 0; f < FF; f++) {
      float x = xt[f];
      gir += Wihf[h * FF + f] * x;
      giz += Wihf[(HH + h) * FF + f] * x;
      gin += Wihf[(2 * HH + h) * FF + f] * x;
    }
    float ghr = bhhf[h], ghz = bhhf[HH + h], ghn = bhhf[2 * HH + h];
    for (int f = 0; f < HH; f++) {
      float x = hp[f];
      ghr += Whhf[h * HH + f] * x;
      ghz += Whhf[(HH + h) * HH + f] * x;
      ghn += Whhf[(2 * HH + h) * HH + f] * x;
    }
    float r = 1.f / (1.f + expf(-(gir + ghr)));
    float z = 1.f / (1.f + expf(-(giz + ghz)));
    float n = tanhf(gin + r * ghn);
    float h2 = (1.f - z) * n + z * hp[h];
    __syncthreads();
    hp[h] = h2;
    __syncthreads();
    hreg = h2;
  }
  const float fwd_last = hreg;

  // backward: bwd[:,0] is exactly one step on x[:, L-1] from h0=0
  {
    const int node = paths[s * LL + (LL - 1)];
    xt[h] = feats[(size_t)node * FF + h];
    __syncthreads();
    float gir = bihb[h], giz = bihb[HH + h], gin = bihb[2 * HH + h];
    for (int f = 0; f < FF; f++) {
      float x = xt[f];
      gir += Wihb[h * FF + f] * x;
      giz += Wihb[(HH + h) * FF + f] * x;
      gin += Wihb[(2 * HH + h) * FF + f] * x;
    }
    float ghr = bhhb[h], ghz = bhhb[HH + h], ghn = bhhb[2 * HH + h]; // h0 = 0
    float r = 1.f / (1.f + expf(-(gir + ghr)));
    float z = 1.f / (1.f + expf(-(giz + ghz)));
    float n = tanhf(gin + r * ghn);
    float h2 = (1.f - z) * n;            // + z*0
    pf[(size_t)s * 2 * HH + h] = fwd_last;
    pf[(size_t)s * 2 * HH + HH + h] = h2;
    out_pf[(size_t)s * 2 * HH + h] = fwd_last;
    out_pf[(size_t)s * 2 * HH + HH + h] = h2;
  }
}

// =================================================================== K6 =====
// path scoring MLP + sigmoid + softmax over K; also emit paths as floats
__global__ void __launch_bounds__(64)
k_score(const float* __restrict__ pf, const float* __restrict__ W1,
        const float* __restrict__ b1, const float* __restrict__ W2,
        const float* __restrict__ b2, const int* __restrict__ paths,
        float* __restrict__ out) {
  const int b = blockIdx.x, tid = threadIdx.x;
  __shared__ float red[64];
  __shared__ float sc[KK];
  for (int k = 0; k < KK; k++) {
    const float* row = pf + (size_t)(b * KK + k) * (2 * HH);
    float hh = b1[tid];
    for (int j = 0; j < 2 * HH; j++) hh += W1[tid * (2 * HH) + j] * row[j];
    hh = fmaxf(hh, 0.f);
    red[tid] = hh * W2[tid];
    __syncthreads();
    for (int o = 32; o >= 1; o >>= 1) { if (tid < o) red[tid] += red[tid + o]; __syncthreads(); }
    if (tid == 0) sc[k] = 1.f / (1.f + expf(-(red[0] + b2[0])));
    __syncthreads();
  }
  if (tid == 0) {
    float mx = sc[0];
    for (int k = 1; k < KK; k++) mx = fmaxf(mx, sc[k]);
    float e[KK], sum = 0.f;
    for (int k = 0; k < KK; k++) { e[k] = expf(sc[k] - mx); sum += e[k]; }
    for (int k = 0; k < KK; k++) out[BB * KK * LL + b * KK + k] = e[k] / sum;
  }
  if (tid < KK * LL) out[b * KK * LL + tid] = (float)paths[b * KK * LL + tid];
}

// ============================================================= launcher =====
extern "C" void kernel_launch(void* const* d_in, const int* in_sizes, int n_in,
                              void* d_out, int out_size, void* d_ws, size_t ws_size,
                              hipStream_t stream) {
  (void)in_sizes; (void)n_in; (void)out_size; (void)ws_size;
  const float* nf    = (const float*)d_in[0];   // (B,T,N,F)
  const float* adj   = (const float*)d_in[1];   // (B,N,N)
  const float* W_sp1 = (const float*)d_in[2];
  const float* b_sp1 = (const float*)d_in[3];
  const float* W_sp2 = (const float*)d_in[4];
  const float* b_sp2 = (const float*)d_in[5];
  const float* Wih_f = (const float*)d_in[6];
  const float* Whh_f = (const float*)d_in[7];
  const float* bih_f = (const float*)d_in[8];
  const float* bhh_f = (const float*)d_in[9];
  const float* Wih_b = (const float*)d_in[10];
  const float* Whh_b = (const float*)d_in[11];
  const float* bih_b = (const float*)d_in[12];
  const float* bhh_b = (const float*)d_in[13];
  const float* W_ps1 = (const float*)d_in[14];
  const float* b_ps1 = (const float*)d_in[15];
  const float* W_ps2 = (const float*)d_in[16];
  const float* b_ps2 = (const float*)d_in[17];

  float* out = (float*)d_out;                   // [paths(320) | weights(64) | pf(8192)]
  float* out_pf = out + BB * KK * LL + BB * KK;

  // workspace layout
  float* fch    = (float*)d_ws;                 // B*N
  float* scaled = fch + BB * NN;                // B*N
  int*   srcIdx = (int*)(scaled + BB * NN);     // B*K
  int*   paths  = srcIdx + BB * KK;             // B*K*L
  float* pf     = (float*)(paths + BB * KK * LL); // B*K*2H

  // K1: fchange (bandwidth-bound pass over 100 MB)
  k_fchange<<<(BB * NN) / 8, 256, 0, stream>>>(nf, fch);
  // K2: WMMA bf16 source MLP, fused epilogue -> scaled logits
  k_srcmlp<<<(BB * NN / 16) / 4, 128, 0, stream>>>(nf, W_sp1, b_sp1, W_sp2, b_sp2,
                                                   fch, scaled);
  // K3: per-batch top-K source selection
  k_topk<<<BB, 256, 0, stream>>>(scaled, srcIdx);
  // K4: biased random walks (Threefry RNG)
  k_walk<<<BB * KK, 256, 0, stream>>>(adj, nf, fch, srcIdx, paths);
  // K5: bidirectional GRU encodings -> pf
  k_gru<<<BB * KK, 64, 0, stream>>>(nf, paths, Wih_f, Whh_f, bih_f, bhh_f,
                                    Wih_b, Whh_b, bih_b, bhh_b, pf, out_pf);
  // K6: path scoring + softmax weights + paths-as-float
  k_score<<<BB, 64, 0, stream>>>(pf, W_ps1, b_ps1, W_ps2, b_ps2, paths, out);
}
